// GCViTLayer_25915832664736
// MI455X (gfx1250) — matrix operand order repaired and checked by reading, not compiled, not run on hardware
//
#include <hip/hip_runtime.h>
#include <math.h>

// ---------------------------------------------------------------------------
// GC-ViT layer for MI455X (gfx1250, wave32, WMMA).
// All dense GEMMs use v_wmma_f32_16x16x32_bf16 (bf16 in, f32 accumulate).
// ---------------------------------------------------------------------------

typedef __attribute__((ext_vector_type(16))) __bf16          v16bf;
typedef __attribute__((ext_vector_type(8)))  float           v8f;
typedef __attribute__((ext_vector_type(8)))  unsigned short  u16x8;
typedef __attribute__((ext_vector_type(16))) unsigned short  u16x16;

union BF16Frag {
    v16bf  v;
    unsigned short u[16];
    u16x8  h[2];
    u16x16 w;
};

__device__ __forceinline__ unsigned short f2bf(float f) {
    unsigned int x = __float_as_uint(f);
    unsigned int r = x + 0x7FFFu + ((x >> 16) & 1u);   // round-to-nearest-even
    return (unsigned short)(r >> 16);
}

__device__ __forceinline__ float gelu_f(float x) {
    return 0.5f * x * (1.0f + erff(x * 0.70710678118654752f));
}

// ---------------------------------------------------------------------------
// Generic WMMA GEMM: C[M,N] = act(A[M,K] (*a_scale) @ B + bias) (+resid)
//   BKN == 0 : B stored (N,K) row-major  (i.e.  x @ W.T, W = (N,K))
//   BKN == 1 : B stored (K,N) row-major  (pointwise conv weights)
// One wave computes one 16x16 output tile; K stepped by 32 (K % 32 == 0,
// M % 16 == 0, N % 16 == 0 for every call site).
// All flags are template parameters -> no divergent runtime branches.
// ---------------------------------------------------------------------------
template<int BKN, int ACT, int SCALE, int BIAS, int RES>
__global__ __launch_bounds__(128)
void gemm_t(const float* __restrict__ A, const float* __restrict__ Bw,
            const float* __restrict__ bias, const float* __restrict__ resid,
            const float* __restrict__ a_scale, int a_div,
            float* __restrict__ C, int M, int N, int K)
{
    const int lane = threadIdx.x & 31;
    const int wave = threadIdx.x >> 5;
    const int tiles_n = N >> 4;
    const long long tiles_m = (long long)(M >> 4);
    const long long tile = (long long)blockIdx.x * 4 + wave;
    if (tile >= tiles_m * tiles_n) return;
    const int  tn = (int)(tile % tiles_n);
    const long long tm = tile / tiles_n;
    const int l15 = lane & 15;
    const int hi  = (lane >> 4) & 1;

    const long long m_a = tm * 16 + l15;            // row this lane feeds into A
    const int       n_b = tn * 16 + l15;            // column this lane feeds into B
    const float* arow = A + m_a * (long long)K;
    const float* srow = SCALE ? (a_scale + (m_a / a_div) * (long long)K) : (const float*)0;

    v8f acc = {};
    for (int k0 = 0; k0 < K; k0 += 32) {
        BF16Frag a, b;
        // A fragment: lane holds row m_a, 16 consecutive K values (k0 + hi*16)
#pragma unroll
        for (int g = 0; g < 2; ++g) {
            const int base = k0 + hi * 16 + g * 8;
            float av[8];
            *(float4*)&av[0] = *(const float4*)(arow + base);
            *(float4*)&av[4] = *(const float4*)(arow + base + 4);
            if (SCALE) {
                float sv[8];
                *(float4*)&sv[0] = *(const float4*)(srow + base);
                *(float4*)&sv[4] = *(const float4*)(srow + base + 4);
#pragma unroll
                for (int j = 0; j < 8; ++j) av[j] *= sv[j];
            }
#pragma unroll
            for (int j = 0; j < 8; ++j) a.u[g * 8 + j] = f2bf(av[j]);
        }
        // B fragment: lane holds column n_b, 16 consecutive K values
        if (BKN) {
            const int kb = k0 + hi * 16;
#pragma unroll
            for (int i = 0; i < 16; ++i)
                b.u[i] = f2bf(Bw[(long long)(kb + i) * N + n_b]);
        } else {
            const float* brow = Bw + (long long)n_b * K + k0 + hi * 16;
            float bv[16];
            *(float4*)&bv[0]  = *(const float4*)(brow);
            *(float4*)&bv[4]  = *(const float4*)(brow + 4);
            *(float4*)&bv[8]  = *(const float4*)(brow + 8);
            *(float4*)&bv[12] = *(const float4*)(brow + 12);
#pragma unroll
            for (int i = 0; i < 16; ++i) b.u[i] = f2bf(bv[i]);
        }
        acc = __builtin_amdgcn_wmma_f32_16x16x32_bf16(false, a.v, false, b.v,
                                                      (short)0, acc, false, false);
    }

#pragma unroll
    for (int r = 0; r < 8; ++r) {
        const long long m = tm * 16 + r + hi * 8;
        const int       n = tn * 16 + l15;
        float v = acc[r];
        if (BIAS) v += bias[n];
        if (ACT)  v  = gelu_f(v);
        const long long idx = m * (long long)N + n;
        if (RES)  v += resid[idx];
        C[idx] = v;
    }
}

// ---------------------------------------------------------------------------
// Window attention: one workgroup (1 wave) per (window, head).
//   S = scale * q k^T           (WMMA, branchless 64x64 padded store)
//   S += rpb bias, softmax      (per-row sweep, bias index without division)
//   O = P v                     (WMMA, V staged transposed for vector loads)
// ---------------------------------------------------------------------------
__global__ __launch_bounds__(32)
void attn_window(const float* __restrict__ qsrc, long long q_wstride, int q_tstride, int q_wdiv,
                 const float* __restrict__ ksrc, long long kv_wstride, int kv_tstride, int v_off,
                 const float* __restrict__ rpb, float* __restrict__ out)
{
    __shared__ unsigned short sq[64][32];
    __shared__ unsigned short sk[64][32];
    __shared__ unsigned short svT[32][64];   // transposed: [d][t]
    __shared__ float          sS[64][64];
    __shared__ unsigned short sP[64][64];

    const int wh  = blockIdx.x;          // window*3 + head
    const int w   = wh / 3;
    const int h   = wh % 3;
    const int tid = threadIdx.x;
    const float scale = 0.17677669529663687f;   // 1/sqrt(32)

    const float* qb = qsrc + (long long)(w / q_wdiv) * q_wstride + h * 32;
    const float* kb = ksrc + (long long)w * kv_wstride + h * 32;
    const float* vb = kb + v_off;

    for (int e = tid; e < 64 * 32; e += 32) {
        const int t = e >> 5, d = e & 31;
        float qv = 0.f, kv = 0.f, vv = 0.f;
        if (t < 49) {
            qv = qb[(long long)t * q_tstride + d];
            kv = kb[(long long)t * kv_tstride + d];
            vv = vb[(long long)t * kv_tstride + d];
        }
        sq[t][d]  = f2bf(qv);
        sk[t][d]  = f2bf(kv);
        svT[d][t] = f2bf(vv);
    }
    __syncthreads();

    const int l15 = tid & 15;
    const int hi  = tid >> 4;

    // ---- S = scale * q @ k^T (K = 32, one WMMA per 16x16 tile) ----
    for (int mt = 0; mt < 4; ++mt) {
        for (int nt = 0; nt < 4; ++nt) {
            BF16Frag a, b;
#pragma unroll
            for (int g = 0; g < 2; ++g)
                a.h[g] = *(const u16x8*)&sq[mt * 16 + l15][hi * 16 + g * 8];
            b.w = *(const u16x16*)&sk[nt * 16 + l15][hi * 16];
            v8f c = {};
            c = __builtin_amdgcn_wmma_f32_16x16x32_bf16(false, a.v, false, b.v,
                                                        (short)0, c, false, false);
#pragma unroll
            for (int r = 0; r < 8; ++r)
                sS[mt * 16 + r + hi * 8][nt * 16 + l15] = c[r] * scale;
        }
    }
    __syncthreads();

    // ---- bias + softmax rows, write bf16 P zero-padded to 64x64 ----
    for (int m = tid; m < 64; m += 32) {
        if (m < 49) {
            const int yi = m / 7, xi = m % 7;
            const float* rb = rpb + ((yi + 6) * 13 + (xi + 6)) * 3 + h;
            float mx = -1e30f;
            {
                int n = 0;
                for (int yj = 0; yj < 7; ++yj)
                    for (int xj = 0; xj < 7; ++xj, ++n) {
                        const float v = sS[m][n] + rb[-(yj * 13 + xj) * 3];
                        sS[m][n] = v;
                        mx = fmaxf(mx, v);
                    }
            }
            float s = 0.f;
            for (int n = 0; n < 49; ++n) s += __expf(sS[m][n] - mx);
            const float inv = 1.f / s;
            for (int n = 0; n < 64; ++n)
                sP[m][n] = (n < 49) ? f2bf(__expf(sS[m][n] - mx) * inv)
                                    : (unsigned short)0;
        } else {
            for (int n = 0; n < 64; ++n) sP[m][n] = 0;
        }
    }
    __syncthreads();

    // ---- O = P @ V (K = 64 -> two WMMA steps) ----
    for (int mt = 0; mt < 4; ++mt) {
        for (int nt = 0; nt < 2; ++nt) {
            v8f c = {};
            for (int kt = 0; kt < 2; ++kt) {
                BF16Frag a, b;
#pragma unroll
                for (int g = 0; g < 2; ++g)
                    a.h[g] = *(const u16x8*)&sP[mt * 16 + l15][kt * 32 + hi * 16 + g * 8];
                b.w = *(const u16x16*)&svT[nt * 16 + l15][kt * 32 + hi * 16];
                c = __builtin_amdgcn_wmma_f32_16x16x32_bf16(false, a.v, false, b.v,
                                                            (short)0, c, false, false);
            }
#pragma unroll
            for (int r = 0; r < 8; ++r) {
                const int m = mt * 16 + r + hi * 8;
                if (m < 49)
                    out[((long long)w * 49 + m) * 96 + h * 32 + nt * 16 + l15] = c[r];
            }
        }
    }
}

// ---------------------------------------------------------------------------
// Elementwise / helper kernels
// ---------------------------------------------------------------------------
__global__ void ln_kernel(const float* __restrict__ in, const float* __restrict__ g,
                          const float* __restrict__ b, float* __restrict__ out,
                          long long ntok, int C)
{
    const long long t = (long long)blockIdx.x * blockDim.x + threadIdx.x;
    if (t >= ntok) return;
    const float* p = in + t * C;
    float s = 0.f;
    for (int c = 0; c < C; ++c) s += p[c];
    const float mean = s / C;
    float v = 0.f;
    for (int c = 0; c < C; ++c) { const float d = p[c] - mean; v += d * d; }
    const float inv = rsqrtf(v / C + 1e-5f);
    float* o = out + t * C;
    for (int c = 0; c < C; ++c) o[c] = (p[c] - mean) * inv * g[c] + b[c];
}

__global__ void dwconv_gelu(const float* __restrict__ in, const float* __restrict__ w,
                            float* __restrict__ out, int Bn, int H, int W)
{
    const long long idx = (long long)blockIdx.x * blockDim.x + threadIdx.x;
    const long long total = (long long)Bn * H * W * 96;
    if (idx >= total) return;
    const int c = (int)(idx % 96);
    long long p = idx / 96;
    const int x = (int)(p % W); p /= W;
    const int y = (int)(p % H);
    const int b = (int)(p / H);
    float s = 0.f;
    for (int ky = 0; ky < 3; ++ky) {
        const int yy = y + ky - 1;
        if (yy < 0 || yy >= H) continue;
        for (int kx = 0; kx < 3; ++kx) {
            const int xx = x + kx - 1;
            if (xx < 0 || xx >= W) continue;
            s += in[(((long long)b * H + yy) * W + xx) * 96 + c] * w[(ky * 3 + kx) * 96 + c];
        }
    }
    out[idx] = gelu_f(s);
}

__global__ void chan_mean(const float* __restrict__ in, float* __restrict__ mean, int HW)
{
    __shared__ float red[128];
    const int bc = blockIdx.x;               // b*96 + c
    const int b = bc / 96, c = bc % 96;
    const float* p = in + (long long)b * HW * 96 + c;
    float s = 0.f;
    for (int i = threadIdx.x; i < HW; i += 128) s += p[(long long)i * 96];
    red[threadIdx.x] = s;
    __syncthreads();
    for (int off = 64; off; off >>= 1) {
        if ((int)threadIdx.x < off) red[threadIdx.x] += red[threadIdx.x + off];
        __syncthreads();
    }
    if (threadIdx.x == 0) mean[bc] = red[0] / HW;
}

__global__ void se_mlp(const float* __restrict__ mean, const float* __restrict__ w1,
                       const float* __restrict__ w2, float* __restrict__ y)
{
    __shared__ float sm[96];
    __shared__ float sh[24];
    const int b = blockIdx.x, t = threadIdx.x;
    sm[t] = mean[b * 96 + t];
    __syncthreads();
    if (t < 24) {
        float s = 0.f;
        for (int c = 0; c < 96; ++c) s += sm[c] * w1[t * 96 + c];
        sh[t] = gelu_f(s);
    }
    __syncthreads();
    float s = 0.f;
    for (int j = 0; j < 24; ++j) s += sh[j] * w2[t * 24 + j];
    y[b * 96 + t] = 1.f / (1.f + __expf(-s));
}

__global__ void add_maxpool(const float* __restrict__ g, const float* __restrict__ t2,
                            float* __restrict__ out, int Bn, int H)
{
    const int Ho = H / 2;
    const long long idx = (long long)blockIdx.x * blockDim.x + threadIdx.x;
    const long long total = (long long)Bn * Ho * Ho * 96;
    if (idx >= total) return;
    const int c = (int)(idx % 96);
    long long p = idx / 96;
    const int ox = (int)(p % Ho); p /= Ho;
    const int oy = (int)(p % Ho);
    const int b  = (int)(p / Ho);
    float m = -3.4e38f;
    for (int ky = 0; ky < 3; ++ky) {
        const int yy = 2 * oy - 1 + ky;
        if (yy < 0 || yy >= H) continue;
        for (int kx = 0; kx < 3; ++kx) {
            const int xx = 2 * ox - 1 + kx;
            if (xx < 0 || xx >= H) continue;
            const long long si = (((long long)b * H + yy) * H + xx) * 96 + c;
            m = fmaxf(m, g[si] + t2[si]);
        }
    }
    out[idx] = m;
}

__global__ void red_conv(const float* __restrict__ in, const float* __restrict__ w,
                         float* __restrict__ out, int Bn)
{
    // in (B,56,56,96), w (3,3,96,192), out (B,28,28,192), stride 2, pad 1
    const long long idx = (long long)blockIdx.x * blockDim.x + threadIdx.x;
    const long long total = (long long)Bn * 28 * 28 * 192;
    if (idx >= total) return;
    const int co = (int)(idx % 192);
    long long p = idx / 192;
    const int ox = (int)(p % 28); p /= 28;
    const int oy = (int)(p % 28);
    const int b  = (int)(p / 28);
    float s = 0.f;
    for (int ky = 0; ky < 3; ++ky) {
        const int yy = 2 * oy - 1 + ky;
        if (yy < 0 || yy >= 56) continue;
        for (int kx = 0; kx < 3; ++kx) {
            const int xx = 2 * ox - 1 + kx;
            if (xx < 0 || xx >= 56) continue;
            const float* ip = in + (((long long)b * 56 + yy) * 56 + xx) * 96;
            const float* wp = w + (long long)((ky * 3 + kx) * 96) * 192 + co;
            for (int ci = 0; ci < 96; ++ci) s += ip[ci] * wp[(long long)ci * 192];
        }
    }
    out[idx] = s;
}

__global__ void win_part(const float* __restrict__ x, float* __restrict__ xw)
{
    const long long idx = (long long)blockIdx.x * blockDim.x + threadIdx.x;
    const long long total = (long long)2048 * 49 * 96;
    if (idx >= total) return;
    const int c = (int)(idx % 96);
    long long p = idx / 96;
    const int t = (int)(p % 49);
    const long long w = p / 49;
    const int b  = (int)(w >> 6);
    const int wy = (int)((w >> 3) & 7);
    const int wx = (int)(w & 7);
    const int ty = t / 7, tx = t % 7;
    xw[idx] = x[(((long long)b * 56 + wy * 7 + ty) * 56 + wx * 7 + tx) * 96 + c];
}

__global__ void win_rev(const float* __restrict__ xw, float* __restrict__ x)
{
    const long long idx = (long long)blockIdx.x * blockDim.x + threadIdx.x;
    const long long total = (long long)32 * 56 * 56 * 96;
    if (idx >= total) return;
    const int c = (int)(idx % 96);
    long long p = idx / 96;
    const int xx = (int)(p % 56); p /= 56;
    const int yy = (int)(p % 56);
    const int b  = (int)(p / 56);
    const int wy = yy / 7, ty = yy % 7, wx = xx / 7, tx = xx % 7;
    const long long w = (long long)b * 64 + wy * 8 + wx;
    x[idx] = xw[(w * 49 + ty * 7 + tx) * 96 + c];
}

// ---------------------------------------------------------------------------
// Host-side launcher
// ---------------------------------------------------------------------------
static inline unsigned cdiv(long long a, long long b) { return (unsigned)((a + b - 1) / b); }

// Compile-time dispatch over the GEMM variants actually used.
static void launch_gemm(const float* A, const float* Bw, const float* bias,
                        const float* resid, const float* ascale, int adiv,
                        float* C, long long M, int N, int K,
                        int bkn, int act, hipStream_t stream)
{
    const unsigned blocks = cdiv((M / 16) * (N / 16), 4);
#define GL(BKN, ACT, SC, BI, RE)                                              \
    gemm_t<BKN, ACT, SC, BI, RE><<<blocks, 128, 0, stream>>>(                 \
        A, Bw, bias, resid, ascale, adiv, C, (int)M, N, K)
    if (bkn == 0) {
        if (act == 1)            GL(0, 1, 0, 1, 0);   // fc1 (+bias, GELU)
        else if (resid)          GL(0, 0, 0, 1, 1);   // proj / fc2 (+bias, +resid)
        else                     GL(0, 0, 0, 1, 0);   // qkv (+bias)
    } else {
        if (resid)               GL(1, 0, 1, 0, 1);   // ds pointwise (SE scale, +resid)
        else                     GL(1, 0, 1, 0, 0);   // qgen pointwise (SE scale)
    }
#undef GL
}

extern "C" void kernel_launch(void* const* d_in, const int* in_sizes, int n_in,
                              void* d_out, int out_size, void* d_ws, size_t ws_size,
                              hipStream_t stream)
{
    (void)in_sizes; (void)n_in; (void)out_size; (void)ws_size;
    const float* F[48];
    for (int i = 0; i < 48; ++i) F[i] = (const float*)d_in[i];

    // JAX pytree (sorted-key) flatten order:
    const float* x_in = F[0];
    // blocks[i]: attn{proj_b,proj_w,qkv_b,qkv_w,rpb}, fc1_b, fc1_w, fc2_b, fc2_w, n1_b, n1_g, n2_b, n2_g
    const int BOF[2] = {1, 14};
    // ds: feat{dw,pw,se1,se2}, n1_b, n1_g, n2_b, n2_g, red
    const float* ds_dw  = F[27]; const float* ds_pw  = F[28];
    const float* ds_se1 = F[29]; const float* ds_se2 = F[30];
    const float* ds_n1b = F[31]; const float* ds_n1g = F[32];
    const float* ds_n2b = F[33]; const float* ds_n2g = F[34];
    const float* ds_red = F[35];
    // qgen[s]: dw, pw, se1, se2 at 36 + 4*s

    // Workspace layout (floats)
    float* ws = (float*)d_ws;
    const long long SZ_FULL = 32LL * 56 * 56 * 96;       // 9,633,792
    float* bufA   = ws;
    float* bufB   = bufA + SZ_FULL;
    float* bufC   = bufB + SZ_FULL;
    float* bufBig = bufC + SZ_FULL;                      // 100352*384
    float* bufG1  = bufBig + 100352LL * 384;
    float* bufG2  = bufG1 + 32LL * 28 * 28 * 96;
    float* bufG3  = bufG2 + 32LL * 14 * 14 * 96;
    float* bufMean = bufG3 + 32LL * 7 * 7 * 96;
    float* bufY    = bufMean + 32 * 96;

    const long long NTOK = 100352;   // 2048 windows * 49 tokens
    const float* ZNULL = (const float*)0;

    // ---------- Phase 1: GlobalQueryGen (56 -> 28 -> 14 -> 7) ----------
    {
        const float* gin = x_in;
        float* gouts[3] = {bufG1, bufG2, bufG3};
        int Hs[3] = {56, 28, 14};
        for (int s = 0; s < 3; ++s) {
            const int H = Hs[s];
            const long long pix = (long long)H * H;
            const float* dw  = F[36 + 4 * s + 0];
            const float* pw  = F[36 + 4 * s + 1];
            const float* se1 = F[36 + 4 * s + 2];
            const float* se2 = F[36 + 4 * s + 3];
            const long long tot = 32LL * pix * 96;
            dwconv_gelu<<<cdiv(tot, 256), 256, 0, stream>>>(gin, dw, bufB, 32, H, H);
            chan_mean<<<32 * 96, 128, 0, stream>>>(bufB, bufMean, (int)pix);
            se_mlp<<<32, 96, 0, stream>>>(bufMean, se1, se2, bufY);
            launch_gemm(bufB, pw, ZNULL, ZNULL, bufY, (int)pix,
                        bufC, 32LL * pix, 96, 96, 1, 0, stream);
            const long long pot = 32LL * (pix / 4) * 96;
            add_maxpool<<<cdiv(pot, 256), 256, 0, stream>>>(gin, bufC, gouts[s], 32, H);
            gin = gouts[s];
        }
    }

    // ---------- Phase 2: window partition + two transformer blocks ----------
    win_part<<<cdiv(NTOK * 96, 256), 256, 0, stream>>>(x_in, bufA);

    for (int blk = 0; blk < 2; ++blk) {
        const int o = BOF[blk];
        const float* proj_b = F[o + 0]; const float* proj_w = F[o + 1];
        const float* qkv_b  = F[o + 2]; const float* qkv_w  = F[o + 3];
        const float* rpb    = F[o + 4];
        const float* fc1_b  = F[o + 5]; const float* fc1_w  = F[o + 6];
        const float* fc2_b  = F[o + 7]; const float* fc2_w  = F[o + 8];
        const float* n1_b   = F[o + 9]; const float* n1_g   = F[o + 10];
        const float* n2_b   = F[o + 11]; const float* n2_g  = F[o + 12];

        // LN1
        ln_kernel<<<cdiv(NTOK, 256), 256, 0, stream>>>(bufA, n1_g, n1_b, bufB, NTOK, 96);

        if (blk == 0) {
            // local qkv: N = 288
            launch_gemm(bufB, qkv_w, qkv_b, ZNULL, ZNULL, 1,
                        bufBig, NTOK, 288, 96, 0, 0, stream);
            attn_window<<<2048 * 3, 32, 0, stream>>>(
                bufBig, 49LL * 288, 288, 1,          // q = qkv[:, 0:96]
                bufBig + 96, 49LL * 288, 288, 96,    // k = qkv[:, 96:192], v = +96
                rpb, bufC);
        } else {
            // global query: kv only, N = 192; q from bufG3 (b,49,96)
            launch_gemm(bufB, qkv_w, qkv_b, ZNULL, ZNULL, 1,
                        bufBig, NTOK, 192, 96, 0, 0, stream);
            attn_window<<<2048 * 3, 32, 0, stream>>>(
                bufG3, 49LL * 96, 96, 64,            // q[b] broadcast over 64 windows
                bufBig, 49LL * 192, 192, 96,         // k = kv[:, 0:96], v = +96
                rpb, bufC);
        }

        // proj + residual (in place on bufA)
        launch_gemm(bufC, proj_w, proj_b, bufA, ZNULL, 1,
                    bufA, NTOK, 96, 96, 0, 0, stream);
        // LN2 + MLP
        ln_kernel<<<cdiv(NTOK, 256), 256, 0, stream>>>(bufA, n2_g, n2_b, bufB, NTOK, 96);
        launch_gemm(bufB, fc1_w, fc1_b, ZNULL, ZNULL, 1,
                    bufBig, NTOK, 384, 96, 0, 1, stream);        // GELU
        launch_gemm(bufBig, fc2_w, fc2_b, bufA, ZNULL, 1,
                    bufA, NTOK, 96, 384, 0, 0, stream);
    }

    // ---------- Phase 3: window reverse + ReduceSize ----------
    win_rev<<<cdiv(SZ_FULL, 256), 256, 0, stream>>>(bufA, bufC);
    ln_kernel<<<cdiv(NTOK, 256), 256, 0, stream>>>(bufC, ds_n1g, ds_n1b, bufB, NTOK, 96);
    dwconv_gelu<<<cdiv(SZ_FULL, 256), 256, 0, stream>>>(bufB, ds_dw, bufA, 32, 56, 56);
    chan_mean<<<32 * 96, 128, 0, stream>>>(bufA, bufMean, 56 * 56);
    se_mlp<<<32, 96, 0, stream>>>(bufMean, ds_se1, ds_se2, bufY);
    // pointwise conv + residual(h) -> h2 in bufBig
    launch_gemm(bufA, ds_pw, ZNULL, bufB, bufY, 56 * 56,
                bufBig, NTOK, 96, 96, 1, 1, stream);
    // strided 3x3 conv 96 -> 192 (stride 2, pad 1) -> bufA (25088 x 192)
    const long long redtot = 32LL * 28 * 28 * 192;
    red_conv<<<cdiv(redtot, 256), 256, 0, stream>>>(bufBig, ds_red, bufA, 32);
    // final LN over 192 channels -> d_out
    ln_kernel<<<cdiv(32LL * 28 * 28, 256), 256, 0, stream>>>(
        bufA, ds_n2g, ds_n2b, (float*)d_out, 32LL * 28 * 28, 192);
}
